// ComplexLSTMLayer_66374424592995
// MI455X (gfx1250) — matrix operand order repaired and checked by reading, not compiled, not run on hardware
//
#include <hip/hip_runtime.h>
#include <math.h>

// ---------------------------------------------------------------------------
// ComplexLSTM (B=32,S=512,I=H=512) for MI455X (gfx1250, wave32, WMMA).
//
// Phase 1: X[16384x2560] = [z_r|z_i][16384x1024] @ WxT-packed(bf16) + bias
//          via v_wmma_f32_16x16x32_bf16 (f32 accum), one pass, ~86 GFLOP.
// Phase 2: 512 serial recurrence steps on a 16-workgroup cluster:
//          - each WGP pins a 1024x160 bf16 slice of the packed recurrent
//            weight in 320 KB LDS, loaded ONCE via TDM tensor_load_to_lds
//            (TENSORcnt-tracked), k-contiguous so B-operand ds_load_b128
//            reads are contiguous;
//          - per step: 32x1024x160 bf16 WMMA GEMM (4 waves x 160 wmma),
//            gate math fully wave-local (c-state in registers),
//            64 KB h-state ping-ponged through L2,
//            cluster split-barrier s_barrier_signal/-wait -3 per step.
// ---------------------------------------------------------------------------

typedef __bf16   v16bf __attribute__((ext_vector_type(16)));
typedef float    v8f   __attribute__((ext_vector_type(8)));
typedef unsigned u32x4 __attribute__((ext_vector_type(4)));
typedef int      i32x4 __attribute__((ext_vector_type(4)));
typedef int      i32x8 __attribute__((ext_vector_type(8)));

#define Bz   32
#define Sz   512
#define Hz   512
#define Kc   1024          // 2*H  (concat [h_r|h_i] / [z_r|z_i])
#define NG   2560          // 5*H  (gate cols: i,f,o,c_r,c_i)
#define Mrow 16384         // B*S

// workspace layout (bytes)
#define OFF_WXT   0ull                      // bf16 [NG][Kc]  (n-major, k-contig)
#define OFF_WHS   5242880ull                // bf16 [16][160][Kc] per-WGP slices
#define OFF_BIAS  10485760ull               // f32  [NG]
#define OFF_HBUF  10496000ull               // bf16 2 x [Bz][Kc] ping-pong
#define OFF_ZP    10627072ull               // bf16 [Mrow][Kc]
#define OFF_X     44181504ull               // f32  [Mrow][NG]

#define OUT_SEQ   8388608                   // B*S*H
#define OUT_FIN   16777216                  // 2*B*S*H

union V16 { v16bf v; uint4 q[2]; };

__device__ __forceinline__ float sigmoidf_(float x) {
  return 1.0f / (1.0f + __expf(-x));
}
__device__ __forceinline__ void ctanh_(float r, float i, float& tr, float& ti) {
  float mag = fmaxf(sqrtf(fmaf(r, r, fmaf(i, i, 1e-14f))), 1e-8f);
  float t   = tanhf(mag) / mag;
  tr = r * t;  ti = i * t;
}

// ---------------------------------------------------------------------------
// Pack weights: fold complex sign structure, cast to bf16, transpose to
// n-major/k-contiguous. Rows 0..2559: WxT row n. Rows 2560..5119: WhS row.
// Gate blocks g: 0=i,1=f,2=o,3=c_r,4=c_i.
//   g<4 : k<512 -> +W_real[k][h],  k>=512 -> -W_imag[k-512][h]
//   g=4 : k<512 -> +W_imag[k][h],  k>=512 -> +W_real[k-512][h]
// ---------------------------------------------------------------------------
__global__ void __launch_bounds__(256) pack_weights_k(
    const float* Wzi_r, const float* Wzi_i, const float* Whi_r, const float* Whi_i,
    const float* Wzf_r, const float* Wzf_i, const float* Whf_r, const float* Whf_i,
    const float* Wzc_r, const float* Wzc_i, const float* Whc_r, const float* Whc_i,
    const float* Wzo_r, const float* Wzo_i, const float* Who_r, const float* Who_i,
    __bf16* WxT, __bf16* WhS)
{
  const float* Zr[5] = { Wzi_r, Wzf_r, Wzo_r, Wzc_r, Wzc_r };
  const float* Zi[5] = { Wzi_i, Wzf_i, Wzo_i, Wzc_i, Wzc_i };
  const float* Hr[5] = { Whi_r, Whf_r, Who_r, Whc_r, Whc_r };
  const float* Hi[5] = { Whi_i, Whf_i, Who_i, Whc_i, Whc_i };

  int row = blockIdx.x;
  if (row < NG) {                       // WxT
    int g = row / Hz, h = row % Hz;
    const float* Wr = Zr[g]; const float* Wi = Zi[g];
    for (int it = 0; it < 4; ++it) {
      int k = threadIdx.x + it * 256;
      float v;
      if (g < 4) v = (k < Hz) ? Wr[k * Hz + h] : -Wi[(k - Hz) * Hz + h];
      else       v = (k < Hz) ? Wi[k * Hz + h] :  Wr[(k - Hz) * Hz + h];
      WxT[(size_t)row * Kc + k] = (__bf16)v;
    }
  } else {                              // WhS: per-WGP sliced recurrent weight
    int r2 = row - NG;                  // r2 = w*160 + c,  c = g*32 + j
    int w = r2 / 160, c = r2 % 160;
    int g = c / 32,  j = c % 32;
    int h = w * 32 + j;
    const float* Wr = Hr[g]; const float* Wi = Hi[g];
    for (int it = 0; it < 4; ++it) {
      int k = threadIdx.x + it * 256;
      float v;
      if (g < 4) v = (k < Hz) ? Wr[k * Hz + h] : -Wi[(k - Hz) * Hz + h];
      else       v = (k < Hz) ? Wi[k * Hz + h] :  Wr[(k - Hz) * Hz + h];
      WhS[(size_t)r2 * Kc + k] = (__bf16)v;
    }
  }
}

__global__ void __launch_bounds__(256) pack_bias_k(
    const float* bi, const float* bf, const float* bo,
    const float* bcr, const float* bci, float* bias)
{
  int n = blockIdx.x * 256 + threadIdx.x;           // 10 blocks * 256 = 2560
  int g = n / Hz, h = n % Hz;
  const float* src[5] = { bi, bf, bo, bcr, bci };
  bias[n] = src[g][h];
}

__global__ void __launch_bounds__(256) pack_z_k(
    const float* zr, const float* zi, __bf16* Zp)
{
  int m = blockIdx.x;                                // m = s*32 + b
  int s = m / Bz, b = m % Bz;
  size_t zbase = ((size_t)b * Sz + s) * 512;
  for (int it = 0; it < 4; ++it) {
    int k = threadIdx.x + it * 256;
    float v = (k < Hz) ? zr[zbase + k] : zi[zbase + (k - Hz)];
    Zp[(size_t)m * Kc + k] = (__bf16)v;
  }
}

__global__ void __launch_bounds__(256) init_h_k(unsigned* hbuf)
{
  int i = blockIdx.x * 256 + threadIdx.x;            // 128 blocks: 32768 dwords
  hbuf[i] = 0u;
}

// ---------------------------------------------------------------------------
// Phase-1 GEMM: X = Zp @ Wx + bias.   M=16384, N=2560, K=1024, bf16->f32.
// Block = 256 threads = 8 waves, tile 64(M) x 128(N); wave = 32x32 (2x2 wmma).
// ---------------------------------------------------------------------------
__global__ void __launch_bounds__(256) xgemm_k(
    const __bf16* __restrict__ Zp, const __bf16* __restrict__ WxT,
    const float* __restrict__ bias, float* __restrict__ X)
{
  int mb = blockIdx.x % (Mrow / 64);
  int nb = blockIdx.x / (Mrow / 64);
  int tid = threadIdx.x, lane = tid & 31, q = tid >> 5;
  int wm = q >> 2, wn = q & 3;
  int half = lane >> 4, l16 = lane & 15;
  int mw = mb * 64 + wm * 32;
  int nw = nb * 128 + wn * 32;

  v8f acc[2][2] = {};
  float bs[2];
  bs[0] = bias[nw + l16];
  bs[1] = bias[nw + 16 + l16];

  for (int kk = 0; kk < 32; ++kk) {
    int k0 = kk * 32;
    V16 a[2], bt[2];
#pragma unroll
    for (int mt = 0; mt < 2; ++mt) {
      const __bf16* ap = Zp + (size_t)(mw + mt * 16 + l16) * Kc + k0 + half * 8;
      a[mt].q[0] = *(const uint4*)(ap);
      a[mt].q[1] = *(const uint4*)(ap + 16);
      // near-cache (WGP-scope) prefetch of next K-tile, not SYS-scope
      if (kk + 1 < 32) __builtin_prefetch((const void*)(ap + 32), 0, 3);
    }
#pragma unroll
    for (int nt = 0; nt < 2; ++nt) {
      const __bf16* bp = WxT + (size_t)(nw + nt * 16 + l16) * Kc + k0 + half * 8;
      bt[nt].q[0] = *(const uint4*)(bp);
      bt[nt].q[1] = *(const uint4*)(bp + 16);
    }
#pragma unroll
    for (int mt = 0; mt < 2; ++mt)
#pragma unroll
      for (int nt = 0; nt < 2; ++nt)
        acc[mt][nt] = __builtin_amdgcn_wmma_f32_16x16x32_bf16(
            false, a[mt].v, false, bt[nt].v, 0, acc[mt][nt], false, false);
  }

#pragma unroll
  for (int mt = 0; mt < 2; ++mt)
#pragma unroll
    for (int nt = 0; nt < 2; ++nt)
#pragma unroll
      for (int r = 0; r < 8; ++r) {
        int m = mw + mt * 16 + half * 8 + r;
        int n = nw + nt * 16 + l16;
        X[(size_t)m * NG + n] = acc[mt][nt][r] + bs[nt];
      }
}

// ---------------------------------------------------------------------------
// Phase-2 recurrence. Grid = 16 workgroups (intended as one 16-WGP cluster),
// 128 threads = 4 waves; wave q = (mt,sub) owns batches [16mt,16mt+16) and
// h-columns [w*32+sub*16, +16) for ALL 5 gates -> gate math is wave-local.
// Weight slice (1024x160 bf16 = 320 KB) pinned in LDS via one TDM load.
// ---------------------------------------------------------------------------
__global__ void __launch_bounds__(128) recurrent_k(
    const __bf16* __restrict__ WhS, const float* __restrict__ X,
    __bf16* __restrict__ hbuf, float* __restrict__ out)
{
  extern __shared__ char smem[];                 // 320 KB dynamic LDS
  const __bf16* Wlds = (const __bf16*)smem;

  const int w    = blockIdx.x;
  const int tid  = threadIdx.x;
  const int lane = tid & 31;
  const int q    = tid >> 5;
  const int mt   = q >> 1, sub = q & 1;
  const int half = lane >> 4, l16 = lane & 15;
  const int hcol = w * 32 + sub * 16 + l16;      // this lane's h column
  const int m_a  = mt * 16 + l16;                // A-operand batch row

  // --- TDM: stage this WGP's 1024x160 bf16 weight slice into LDS (one op).
  if (tid < 32) {                                 // wave 0 only (EXEC ignored)
    unsigned long long ga =
        (unsigned long long)(const void*)(WhS + (size_t)w * 160 * Kc);
    u32x4 g0 = { 1u,                               // count=1, gather off
                 0u,                               // lds_addr = dyn-LDS base
                 (unsigned)(ga & 0xFFFFFFFFull),
                 (unsigned)((ga >> 32) & 0x01FFFFFFull) | (2u << 30) }; // type=2
    // group1: data_size=2B; tensor 1024x160, stride 1024; tile 1024x160
    i32x8 g1 = { 0x00010000,
                 (int)(1024u << 16),               // tensor_dim0 = 1024
                 (int)(160u  << 16),               // tensor_dim1 = 160
                 (int)(1024u << 16),               // tile_dim0   = 1024
                 160,                              // tile_dim1=160, tile_dim2=0
                 1024,                             // tensor_dim0_stride
                 0, 0 };
    i32x4 z4 = { 0, 0, 0, 0 };
#if defined(__clang_major__) && (__clang_major__ >= 23)
    i32x8 z8 = { 0, 0, 0, 0, 0, 0, 0, 0 };
    __builtin_amdgcn_tensor_load_to_lds(g0, g1, z4, z4, z8, 0);
#else
    __builtin_amdgcn_tensor_load_to_lds(g0, g1, z4, z4, 0);
#endif
    __builtin_amdgcn_s_wait_tensorcnt(0);
  }
  __syncthreads();

  float cr_s[8] = {}, ci_s[8] = {};

  for (int s = 0; s < Sz; ++s) {
    const __bf16* hc = hbuf + (size_t)(s & 1) * (Bz * Kc);
    __bf16*       hn = hbuf + (size_t)((s + 1) & 1) * (Bz * Kc);

    // acc init = precomputed x-projection (bias already fused in).
    v8f acc[5];
#pragma unroll
    for (int g = 0; g < 5; ++g)
#pragma unroll
      for (int r = 0; r < 8; ++r) {
        int b = mt * 16 + half * 8 + r;
        acc[g][r] = X[((size_t)(s * Bz + b)) * NG + g * Hz + hcol];
      }

    // G += [h_r|h_i] @ WhSlice  (K = 1024, 32 wmma-K steps)
#pragma unroll 4
    for (int kk = 0; kk < 32; ++kk) {
      int k0 = kk * 32;
      V16 a;
      const __bf16* ap = hc + (size_t)m_a * Kc + k0 + half * 8;
      a.q[0] = *(const uint4*)(ap);
      a.q[1] = *(const uint4*)(ap + 16);
#pragma unroll
      for (int g = 0; g < 5; ++g) {
        V16 bt;
        const __bf16* bp =
            Wlds + (size_t)(g * 32 + sub * 16 + l16) * Kc + k0 + half * 8;
        bt.q[0] = *(const uint4*)(bp);
        bt.q[1] = *(const uint4*)(bp + 16);
        acc[g] = __builtin_amdgcn_wmma_f32_16x16x32_bf16(
            false, a.v, false, bt.v, 0, acc[g], false, false);
      }
    }

    // gate nonlinearities + state update (fully lane/register local)
#pragma unroll
    for (int r = 0; r < 8; ++r) {
      int b = mt * 16 + half * 8 + r;
      float it = sigmoidf_(acc[0][r]);
      float ft = sigmoidf_(acc[1][r]);
      float ot = sigmoidf_(acc[2][r]);
      float tr, ti;
      ctanh_(acc[3][r], acc[4][r], tr, ti);
      cr_s[r] = ft * cr_s[r] + it * tr;
      ci_s[r] = ft * ci_s[r] + it * ti;
      float thr, thi;
      ctanh_(cr_s[r], ci_s[r], thr, thi);
      float hr = ot * thr, hi = ot * thi;

      hn[(size_t)b * Kc + hcol]       = (__bf16)hr;   // next-step A operand
      hn[(size_t)b * Kc + Hz + hcol]  = (__bf16)hi;

      size_t o = ((size_t)b * Sz + s) * Hz + hcol;
      out[o]           = hr;
      out[OUT_SEQ + o] = hi;
      if (s == Sz - 1) {
        size_t f = (size_t)b * Hz + hcol;
        out[OUT_FIN + f]         = hr;
        out[OUT_FIN + 16384 + f] = hi;
        out[OUT_FIN + 32768 + f] = cr_s[r];
        out[OUT_FIN + 49152 + f] = ci_s[r];
      }
    }

    // publish h, then cluster-wide split barrier (ISA: one wave/WG signals,
    // all waves wait; NOP if the dispatch is not cluster-configured).
    __threadfence();
    __syncthreads();
    if (tid < 32) asm volatile("s_barrier_signal -3" ::: "memory");
    asm volatile("s_barrier_wait -3" ::: "memory");
    __threadfence();
  }
}

// ---------------------------------------------------------------------------
extern "C" void kernel_launch(void* const* d_in, const int* in_sizes, int n_in,
                              void* d_out, int out_size, void* d_ws, size_t ws_size,
                              hipStream_t stream) {
  (void)in_sizes; (void)n_in; (void)out_size; (void)ws_size;
  char* ws = (char*)d_ws;
  __bf16* WxT  = (__bf16*)(ws + OFF_WXT);
  __bf16* WhS  = (__bf16*)(ws + OFF_WHS);
  float*  bias = (float*) (ws + OFF_BIAS);
  __bf16* hbuf = (__bf16*)(ws + OFF_HBUF);
  __bf16* Zp   = (__bf16*)(ws + OFF_ZP);
  float*  X    = (float*) (ws + OFF_X);

  const float* zr = (const float*)d_in[0];
  const float* zi = (const float*)d_in[1];

  pack_weights_k<<<5120, 256, 0, stream>>>(
      (const float*)d_in[2],  (const float*)d_in[3],  (const float*)d_in[4],  (const float*)d_in[5],
      (const float*)d_in[8],  (const float*)d_in[9],  (const float*)d_in[10], (const float*)d_in[11],
      (const float*)d_in[14], (const float*)d_in[15], (const float*)d_in[16], (const float*)d_in[17],
      (const float*)d_in[20], (const float*)d_in[21], (const float*)d_in[22], (const float*)d_in[23],
      WxT, WhS);
  pack_bias_k<<<10, 256, 0, stream>>>(
      (const float*)d_in[6], (const float*)d_in[12], (const float*)d_in[24],
      (const float*)d_in[18], (const float*)d_in[19], bias);
  pack_z_k<<<Mrow, 256, 0, stream>>>(zr, zi, Zp);
  init_h_k<<<128, 256, 0, stream>>>((unsigned*)hbuf);

  xgemm_k<<<(Mrow / 64) * (NG / 128), 256, 0, stream>>>(Zp, WxT, bias, X);

  recurrent_k<<<16, 128, 320 * 1024, stream>>>(WhS, X, hbuf, (float*)d_out);
}
